// Model5_23922967839350
// MI455X (gfx1250) — matrix operand order repaired and checked by reading, not compiled, not run on hardware
//
#include <hip/hip_runtime.h>
#include <stddef.h>
#include <stdint.h>

// ---------------------------------------------------------------------------
// Types for CDNA5 WMMA (wave32): v_wmma_f32_16x16x32_bf16
// ---------------------------------------------------------------------------
typedef __attribute__((ext_vector_type(16))) __bf16 v16bf;
typedef __attribute__((ext_vector_type(8)))  float  v8f;

#define DEVFN __device__ __forceinline__

DEVFN __bf16 f2bf(float f) { return (__bf16)f; }
DEVFN float  bf2f(__bf16 b) { return (float)b; }
DEVFN float  sigf(float x) { return 1.0f / (1.0f + __expf(-x)); }

// Problem constants (from the reference)
constexpr int Bn    = 4096;
constexpr int Tt    = 100;
constexpr int Ff    = 81;
constexpr int TMc   = 94;          // T - KH + 1
constexpr int Mm    = 100;
constexpr int Pp    = 128;
constexpr int SKIPc = 10;
constexpr int SHc   = 32;
constexpr int RND   = Bn * TMc;    // 385024 rows of H / HU
constexpr int NRG   = Bn * SKIPc;  // 40960 GRU rows
constexpr int KCONV = 567;         // 7*81
constexpr int KOUT1 = 548;         // 100+128+320

// ---------------------------------------------------------------------------
// WMMA inner loop over one K-group staged in LDS.
// All B tiles are loaded into a register array FIRST so the compiler can
// cover the whole batch with a single s_wait_dscnt and issue the NTB
// v_wmma ops back-to-back (XDL pipelining).
// LDS layouts are pre-swizzled to CDNA5 wave32 WMMA VGPR layouts:
//   A chunk: lane L holds row (L&15); half h -> K = base + (L>=16?8:0) + (h<8?h:h+8)
//   B chunk: lane L holds col (L&15); half h -> K = base + (L>=16?8:0) + (h<8?h:h+8)
// ---------------------------------------------------------------------------
template <int GKC, int NTB>
DEVFN void wmma_groups(const __bf16* la, const __bf16* lb, int w, int lane, v8f* acc)
{
#pragma unroll
    for (int kcl = 0; kcl < GKC; ++kcl) {
        v16bf av = *(const v16bf*)(la + (w * GKC + kcl) * 512 + lane * 16);
        v16bf bv[NTB];
#pragma unroll
        for (int nt = 0; nt < NTB; ++nt)
            bv[nt] = *(const v16bf*)(lb + (nt * GKC + kcl) * 512 + lane * 16);
#pragma unroll
        for (int nt = 0; nt < NTB; ++nt)
            acc[nt] = __builtin_amdgcn_wmma_f32_16x16x32_bf16(
                false, av, false, bv[nt], (short)0, acc[nt], false, false);
    }
}

DEVFN int a_k_of(int kcAbs, int L, int h) { return kcAbs * 32 + ((L & 16) ? 8 : 0) + (h < 8 ? h : h + 8); }
DEVFN int b_k_of(int kcAbs, int L, int h) { return kcAbs * 32 + ((L & 16) ? 8 : 0) + (h < 8 ? h : h + 8); }

// ---------------------------------------------------------------------------
// Kernel 1: conv (implicit GEMM, K=567) + bias + ReLU -> Hb (bf16) [RND][100]
// Double-buffered LDS staging: global prefetch of group g+1 overlaps the
// WMMA chain of group g.
// ---------------------------------------------------------------------------
__global__ void k_conv(const float* __restrict__ enc, const float* __restrict__ cw,
                       const float* __restrict__ cb, __bf16* __restrict__ Hb)
{
    constexpr int GKC = 3, NG = 6, NTB = 7, RW = 8;
    __shared__ __attribute__((aligned(32))) __bf16 la[2][RW * GKC * 512];
    __shared__ __attribute__((aligned(32))) __bf16 lb[2][NTB * GKC * 512];
    const int rowBase = blockIdx.x * (RW * 16);
    const int lane = threadIdx.x & 31, w = threadIdx.x >> 5;

    auto fillA = [&](int g, __bf16* dst) {
        for (int idx = threadIdx.x; idx < RW * GKC * 512; idx += blockDim.x) {
            int h = idx & 15, L = (idx >> 4) & 31, kcl = (idx >> 9) % GKC, rt = idx / (GKC * 512);
            int k = a_k_of(g * GKC + kcl, L, h);
            int row = rowBase + rt * 16 + (L & 15);
            float v = 0.0f;
            if (k < KCONV) {
                int b = row / TMc, t = row - b * TMc;
                int kh = k / Ff, f = k - kh * Ff;
                v = enc[((size_t)b * Tt + t + kh) * Ff + f];
            }
            dst[idx] = f2bf(v);
        }
    };
    auto fillB = [&](int g, __bf16* dst) {
        for (int idx = threadIdx.x; idx < NTB * GKC * 512; idx += blockDim.x) {
            int h = idx & 15, L = (idx >> 4) & 31, kcl = (idx >> 9) % GKC, nt = idx / (GKC * 512);
            int k = b_k_of(g * GKC + kcl, L, h);
            int n = nt * 16 + (L & 15);
            float v = (k < KCONV && n < Mm) ? cw[(size_t)n * KCONV + k] : 0.0f;
            dst[idx] = f2bf(v);
        }
    };

    v8f acc[NTB];
#pragma unroll
    for (int i = 0; i < NTB; ++i)
#pragma unroll
        for (int r = 0; r < 8; ++r) acc[i][r] = 0.0f;

    fillA(0, la[0]); fillB(0, lb[0]);
    __syncthreads();
    for (int g = 0; g < NG; ++g) {
        const int cur = g & 1;
        wmma_groups<GKC, NTB>(la[cur], lb[cur], w, lane, acc);
        if (g + 1 < NG) { fillA(g + 1, la[cur ^ 1]); fillB(g + 1, lb[cur ^ 1]); }
        __syncthreads();
    }

    const int rbase = rowBase + w * 16 + ((lane & 16) ? 8 : 0);
    const int nc = lane & 15;
#pragma unroll
    for (int nt = 0; nt < NTB; ++nt) {
        int n = nt * 16 + nc;
        if (n < Mm) {
            float bias = cb[n];
#pragma unroll
            for (int r = 0; r < 8; ++r) {
                float v = acc[nt][r] + bias;
                Hb[(size_t)(rbase + r) * Mm + n] = f2bf(v > 0.0f ? v : 0.0f);
            }
        }
    }
}

// ---------------------------------------------------------------------------
// Kernel 2: HU = H @ Ud^T -> HUb (bf16) [RND][100]
// ---------------------------------------------------------------------------
__global__ void k_hu(const __bf16* __restrict__ Hb, const float* __restrict__ Ud,
                     __bf16* __restrict__ HUb)
{
    constexpr int GKC = 2, NG = 2, NTB = 7, RW = 8;
    __shared__ __attribute__((aligned(32))) __bf16 la[2][RW * GKC * 512];
    __shared__ __attribute__((aligned(32))) __bf16 lb[2][NTB * GKC * 512];
    const int rowBase = blockIdx.x * (RW * 16);
    const int lane = threadIdx.x & 31, w = threadIdx.x >> 5;

    auto fillA = [&](int g, __bf16* dst) {
        for (int idx = threadIdx.x; idx < RW * GKC * 512; idx += blockDim.x) {
            int h = idx & 15, L = (idx >> 4) & 31, kcl = (idx >> 9) % GKC, rt = idx / (GKC * 512);
            int k = a_k_of(g * GKC + kcl, L, h);
            int row = rowBase + rt * 16 + (L & 15);
            dst[idx] = (k < Mm) ? Hb[(size_t)row * Mm + k] : f2bf(0.0f);
        }
    };
    auto fillB = [&](int g, __bf16* dst) {
        for (int idx = threadIdx.x; idx < NTB * GKC * 512; idx += blockDim.x) {
            int h = idx & 15, L = (idx >> 4) & 31, kcl = (idx >> 9) % GKC, nt = idx / (GKC * 512);
            int k = b_k_of(g * GKC + kcl, L, h);
            int n = nt * 16 + (L & 15);
            float v = (k < Mm && n < Mm) ? Ud[(size_t)n * Mm + k] : 0.0f;
            dst[idx] = f2bf(v);
        }
    };

    v8f acc[NTB];
#pragma unroll
    for (int i = 0; i < NTB; ++i)
#pragma unroll
        for (int r = 0; r < 8; ++r) acc[i][r] = 0.0f;

    fillA(0, la[0]); fillB(0, lb[0]);
    __syncthreads();
    for (int g = 0; g < NG; ++g) {
        const int cur = g & 1;
        wmma_groups<GKC, NTB>(la[cur], lb[cur], w, lane, acc);
        if (g + 1 < NG) { fillA(g + 1, la[cur ^ 1]); fillB(g + 1, lb[cur ^ 1]); }
        __syncthreads();
    }

    const int rbase = rowBase + w * 16 + ((lane & 16) ? 8 : 0);
    const int nc = lane & 15;
#pragma unroll
    for (int nt = 0; nt < NTB; ++nt) {
        int n = nt * 16 + nc;
        if (n < Mm) {
#pragma unroll
            for (int r = 0; r < 8; ++r)
                HUb[(size_t)(rbase + r) * Mm + n] = f2bf(acc[nt][r]);
        }
    }
}

// ---------------------------------------------------------------------------
// Kernel 3 (per decoder step): qWd = [d|s] @ Wd^T  (B x 100), f32 out
// ---------------------------------------------------------------------------
__global__ void k_qwd(const float* __restrict__ dv, const float* __restrict__ sv,
                      const float* __restrict__ Wd, float* __restrict__ qWd)
{
    constexpr int GKC = 2, NG = 4, NTB = 7, RW = 8;
    __shared__ __attribute__((aligned(32))) __bf16 la[2][RW * GKC * 512];
    __shared__ __attribute__((aligned(32))) __bf16 lb[2][NTB * GKC * 512];
    const int rowBase = blockIdx.x * (RW * 16);
    const int lane = threadIdx.x & 31, w = threadIdx.x >> 5;

    auto fillA = [&](int g, __bf16* dst) {
        for (int idx = threadIdx.x; idx < RW * GKC * 512; idx += blockDim.x) {
            int h = idx & 15, L = (idx >> 4) & 31, kcl = (idx >> 9) % GKC, rt = idx / (GKC * 512);
            int k = a_k_of(g * GKC + kcl, L, h);
            int row = rowBase + rt * 16 + (L & 15);
            float v = (k < Pp) ? dv[(size_t)row * Pp + k] : sv[(size_t)row * Pp + (k - Pp)];
            dst[idx] = f2bf(v);
        }
    };
    auto fillB = [&](int g, __bf16* dst) {
        for (int idx = threadIdx.x; idx < NTB * GKC * 512; idx += blockDim.x) {
            int h = idx & 15, L = (idx >> 4) & 31, kcl = (idx >> 9) % GKC, nt = idx / (GKC * 512);
            int k = b_k_of(g * GKC + kcl, L, h);
            int n = nt * 16 + (L & 15);
            float v = (n < Mm) ? Wd[(size_t)n * (2 * Pp) + k] : 0.0f;
            dst[idx] = f2bf(v);
        }
    };

    v8f acc[NTB];
#pragma unroll
    for (int i = 0; i < NTB; ++i)
#pragma unroll
        for (int r = 0; r < 8; ++r) acc[i][r] = 0.0f;

    fillA(0, la[0]); fillB(0, lb[0]);
    __syncthreads();
    for (int g = 0; g < NG; ++g) {
        const int cur = g & 1;
        wmma_groups<GKC, NTB>(la[cur], lb[cur], w, lane, acc);
        if (g + 1 < NG) { fillA(g + 1, la[cur ^ 1]); fillB(g + 1, lb[cur ^ 1]); }
        __syncthreads();
    }

    const int rbase = rowBase + w * 16 + ((lane & 16) ? 8 : 0);
    const int nc = lane & 15;
#pragma unroll
    for (int nt = 0; nt < NTB; ++nt) {
        int n = nt * 16 + nc;
        if (n < Mm) {
#pragma unroll
            for (int r = 0; r < 8; ++r)
                qWd[(size_t)(rbase + r) * Mm + n] = acc[nt][r];
        }
    }
}

// ---------------------------------------------------------------------------
// Kernel 4 (per decoder step): attention — scores, softmax, context c
// one block per batch element; HU/H are bf16 (L2-resident across all steps)
// ---------------------------------------------------------------------------
__global__ void k_attn(const float* __restrict__ qWd, const __bf16* __restrict__ HUb,
                       const __bf16* __restrict__ Hb, const float* __restrict__ vd,
                       float* __restrict__ cv)
{
    __shared__ float qk[Mm];
    __shared__ float vl[Mm];
    __shared__ float red[128];
    __shared__ float bt[TMc];
    const int b = blockIdx.x, tid = threadIdx.x;

    if (tid < Mm) { qk[tid] = qWd[(size_t)b * Mm + tid]; vl[tid] = vd[tid]; }
    __syncthreads();

    float s = -1e30f;
    if (tid < TMc) {
        s = 0.0f;
        const __bf16* hu = HUb + ((size_t)b * TMc + tid) * Mm;
        for (int k = 0; k < Mm; ++k) s += vl[k] * tanhf(qk[k] + bf2f(hu[k]));
    }
    red[tid] = s;
    __syncthreads();
    for (int off = 64; off > 0; off >>= 1) {
        if (tid < off) red[tid] = fmaxf(red[tid], red[tid + off]);
        __syncthreads();
    }
    const float mx = red[0];
    __syncthreads();
    float e = (tid < TMc) ? __expf(s - mx) : 0.0f;
    red[tid] = e;
    __syncthreads();
    for (int off = 64; off > 0; off >>= 1) {
        if (tid < off) red[tid] += red[tid + off];
        __syncthreads();
    }
    const float inv = 1.0f / red[0];
    if (tid < TMc) bt[tid] = e * inv;
    __syncthreads();

    if (tid < Mm) {
        float a = 0.0f;
        const __bf16* hp = Hb + (size_t)b * TMc * Mm + tid;
        for (int t = 0; t < TMc; ++t) a += bt[t] * bf2f(hp[(size_t)t * Mm]);
        cv[(size_t)b * Mm + tid] = a;
    }
}

// ---------------------------------------------------------------------------
// Kernel 5 (per decoder step): LSTM gates = [c|d] @ [wih|whh]^T + biases
// N=512 split across blockIdx.y (4 groups of 8 col tiles)
// ---------------------------------------------------------------------------
__global__ void k_gates(const float* __restrict__ cv, const float* __restrict__ dv,
                        const float* __restrict__ wih, const float* __restrict__ whh,
                        const float* __restrict__ bih, const float* __restrict__ bhh,
                        float* __restrict__ gates)
{
    constexpr int GKC = 2, NG = 4, NTB = 8, RW = 8;
    constexpr int KAB = Mm + Pp; // 228
    __shared__ __attribute__((aligned(32))) __bf16 la[2][RW * GKC * 512];
    __shared__ __attribute__((aligned(32))) __bf16 lb[2][NTB * GKC * 512];
    const int rowBase = blockIdx.x * (RW * 16);
    const int nBase = blockIdx.y * (NTB * 16);
    const int lane = threadIdx.x & 31, w = threadIdx.x >> 5;

    auto fillA = [&](int g, __bf16* dst) {
        for (int idx = threadIdx.x; idx < RW * GKC * 512; idx += blockDim.x) {
            int h = idx & 15, L = (idx >> 4) & 31, kcl = (idx >> 9) % GKC, rt = idx / (GKC * 512);
            int k = a_k_of(g * GKC + kcl, L, h);
            int row = rowBase + rt * 16 + (L & 15);
            float v = 0.0f;
            if (k < Mm) v = cv[(size_t)row * Mm + k];
            else if (k < KAB) v = dv[(size_t)row * Pp + (k - Mm)];
            dst[idx] = f2bf(v);
        }
    };
    auto fillB = [&](int g, __bf16* dst) {
        for (int idx = threadIdx.x; idx < NTB * GKC * 512; idx += blockDim.x) {
            int h = idx & 15, L = (idx >> 4) & 31, kcl = (idx >> 9) % GKC, nt = idx / (GKC * 512);
            int k = b_k_of(g * GKC + kcl, L, h);
            int n = nBase + nt * 16 + (L & 15);
            float v = 0.0f;
            if (k < Mm) v = wih[(size_t)n * Mm + k];
            else if (k < KAB) v = whh[(size_t)n * Pp + (k - Mm)];
            dst[idx] = f2bf(v);
        }
    };

    v8f acc[NTB];
#pragma unroll
    for (int i = 0; i < NTB; ++i)
#pragma unroll
        for (int r = 0; r < 8; ++r) acc[i][r] = 0.0f;

    fillA(0, la[0]); fillB(0, lb[0]);
    __syncthreads();
    for (int g = 0; g < NG; ++g) {
        const int cur = g & 1;
        wmma_groups<GKC, NTB>(la[cur], lb[cur], w, lane, acc);
        if (g + 1 < NG) { fillA(g + 1, la[cur ^ 1]); fillB(g + 1, lb[cur ^ 1]); }
        __syncthreads();
    }

    const int rbase = rowBase + w * 16 + ((lane & 16) ? 8 : 0);
    const int nc = lane & 15;
#pragma unroll
    for (int nt = 0; nt < NTB; ++nt) {
        int n = nBase + nt * 16 + nc;
        float bias = bih[n] + bhh[n];
#pragma unroll
        for (int r = 0; r < 8; ++r)
            gates[(size_t)(rbase + r) * (4 * Pp) + n] = acc[nt][r] + bias;
    }
}

// ---------------------------------------------------------------------------
// Kernel 6 (per decoder step): LSTM pointwise update of d, s
// ---------------------------------------------------------------------------
__global__ void k_lstm(const float* __restrict__ gates, float* __restrict__ dv,
                       float* __restrict__ sv)
{
    int i = blockIdx.x * blockDim.x + threadIdx.x;
    if (i >= Bn * Pp) return;
    int b = i >> 7, p = i & 127;
    const float* g = gates + (size_t)b * (4 * Pp);
    float ig = g[p], fg = g[Pp + p], gg = g[2 * Pp + p], og = g[3 * Pp + p];
    float sn = sigf(fg) * sv[i] + sigf(ig) * tanhf(gg);
    sv[i] = sn;
    dv[i] = sigf(og) * tanhf(sn);
}

// ---------------------------------------------------------------------------
// Kernel 7 (per GRU step pt): gates = [x|h] @ blockdiag(wih^T, whh^T) + biases
// N=192 split across blockIdx.y (2 groups of 6 col tiles):
// cols 0..95 = gi, 96..191 = gh (kept separate for the reset gate)
// ---------------------------------------------------------------------------
__global__ void k_gru_gates(const __bf16* __restrict__ Hb, const float* __restrict__ hg,
                            const float* __restrict__ wih, const float* __restrict__ whh,
                            const float* __restrict__ bih, const float* __restrict__ bhh,
                            float* __restrict__ gg, int pt)
{
    constexpr int GKC = 1, NG = 5, NTB = 6, RW = 8;
    constexpr int KAB = Mm + SHc; // 132
    __shared__ __attribute__((aligned(32))) __bf16 la[2][RW * GKC * 512];
    __shared__ __attribute__((aligned(32))) __bf16 lb[2][NTB * GKC * 512];
    const int rowBase = blockIdx.x * (RW * 16);
    const int nBase = blockIdx.y * (NTB * 16);
    const int lane = threadIdx.x & 31, w = threadIdx.x >> 5;

    auto fillA = [&](int g, __bf16* dst) {
        for (int idx = threadIdx.x; idx < RW * GKC * 512; idx += blockDim.x) {
            int h = idx & 15, L = (idx >> 4) & 31, rt = idx / (GKC * 512);
            int k = a_k_of(g, L, h);
            int row = rowBase + rt * 16 + (L & 15);
            __bf16 v = f2bf(0.0f);
            if (k < Mm) {
                int b = row / SKIPc, sk = row - b * SKIPc;
                v = Hb[((size_t)b * TMc + 4 + pt * SKIPc + sk) * Mm + k];
            } else if (k < KAB) {
                v = f2bf(hg[(size_t)row * SHc + (k - Mm)]);
            }
            dst[idx] = v;
        }
    };
    auto fillB = [&](int g, __bf16* dst) {
        for (int idx = threadIdx.x; idx < NTB * GKC * 512; idx += blockDim.x) {
            int h = idx & 15, L = (idx >> 4) & 31, nt = idx / (GKC * 512);
            int k = b_k_of(g, L, h);
            int n = nBase + nt * 16 + (L & 15);
            float v = 0.0f;
            if (n < 3 * SHc) { if (k < Mm) v = wih[(size_t)n * Mm + k]; }
            else             { if (k >= Mm && k < KAB) v = whh[(size_t)(n - 3 * SHc) * SHc + (k - Mm)]; }
            dst[idx] = f2bf(v);
        }
    };

    v8f acc[NTB];
#pragma unroll
    for (int i = 0; i < NTB; ++i)
#pragma unroll
        for (int r = 0; r < 8; ++r) acc[i][r] = 0.0f;

    fillA(0, la[0]); fillB(0, lb[0]);
    __syncthreads();
    for (int g = 0; g < NG; ++g) {
        const int cur = g & 1;
        wmma_groups<GKC, NTB>(la[cur], lb[cur], w, lane, acc);
        if (g + 1 < NG) { fillA(g + 1, la[cur ^ 1]); fillB(g + 1, lb[cur ^ 1]); }
        __syncthreads();
    }

    const int rbase = rowBase + w * 16 + ((lane & 16) ? 8 : 0);
    const int nc = lane & 15;
#pragma unroll
    for (int nt = 0; nt < NTB; ++nt) {
        int n = nBase + nt * 16 + nc;
        float bias = (n < 3 * SHc) ? bih[n] : bhh[n - 3 * SHc];
#pragma unroll
        for (int r = 0; r < 8; ++r)
            gg[(size_t)(rbase + r) * 192 + n] = acc[nt][r] + bias;
    }
}

// ---------------------------------------------------------------------------
// Kernel 8 (per GRU step): pointwise GRU update of h
// ---------------------------------------------------------------------------
__global__ void k_gru_upd(const float* __restrict__ gg, float* __restrict__ hg)
{
    int i = blockIdx.x * blockDim.x + threadIdx.x;
    if (i >= NRG * SHc) return;
    int row = i >> 5, j = i & 31;
    const float* g = gg + (size_t)row * 192;
    float ir = g[j], iz = g[32 + j], in = g[64 + j];
    float hr = g[96 + j], hz = g[128 + j], hn = g[160 + j];
    float h = hg[i];
    float r = sigf(ir + hr), z = sigf(iz + hz);
    float nn = tanhf(in + r * hn);
    hg[i] = (1.0f - z) * nn + z * h;
}

// ---------------------------------------------------------------------------
// Kernel 9: hidden = relu([d | c | h_skip] @ out1_w^T + out1_b)  (B x 128)
// ---------------------------------------------------------------------------
__global__ void k_out1(const float* __restrict__ dv, const float* __restrict__ cv,
                       const float* __restrict__ hg, const float* __restrict__ w1,
                       const float* __restrict__ b1, float* __restrict__ hid)
{
    constexpr int GKC = 3, NG = 6, NTB = 8, RW = 8;
    __shared__ __attribute__((aligned(32))) __bf16 la[2][RW * GKC * 512];
    __shared__ __attribute__((aligned(32))) __bf16 lb[2][NTB * GKC * 512];
    const int rowBase = blockIdx.x * (RW * 16);
    const int lane = threadIdx.x & 31, w = threadIdx.x >> 5;

    auto fillA = [&](int g, __bf16* dst) {
        for (int idx = threadIdx.x; idx < RW * GKC * 512; idx += blockDim.x) {
            int h = idx & 15, L = (idx >> 4) & 31, kcl = (idx >> 9) % GKC, rt = idx / (GKC * 512);
            int k = a_k_of(g * GKC + kcl, L, h);
            int row = rowBase + rt * 16 + (L & 15);
            float v = 0.0f;
            if (k < Pp) v = dv[(size_t)row * Pp + k];
            else if (k < Pp + Mm) v = cv[(size_t)row * Mm + (k - Pp)];
            else if (k < KOUT1) {
                int jj = k - (Pp + Mm);
                v = hg[((size_t)row * SKIPc + (jj >> 5)) * SHc + (jj & 31)];
            }
            dst[idx] = f2bf(v);
        }
    };
    auto fillB = [&](int g, __bf16* dst) {
        for (int idx = threadIdx.x; idx < NTB * GKC * 512; idx += blockDim.x) {
            int h = idx & 15, L = (idx >> 4) & 31, kcl = (idx >> 9) % GKC, nt = idx / (GKC * 512);
            int k = b_k_of(g * GKC + kcl, L, h);
            int n = nt * 16 + (L & 15);
            float v = (k < KOUT1) ? w1[(size_t)n * KOUT1 + k] : 0.0f;
            dst[idx] = f2bf(v);
        }
    };

    v8f acc[NTB];
#pragma unroll
    for (int i = 0; i < NTB; ++i)
#pragma unroll
        for (int r = 0; r < 8; ++r) acc[i][r] = 0.0f;

    fillA(0, la[0]); fillB(0, lb[0]);
    __syncthreads();
    for (int g = 0; g < NG; ++g) {
        const int cur = g & 1;
        wmma_groups<GKC, NTB>(la[cur], lb[cur], w, lane, acc);
        if (g + 1 < NG) { fillA(g + 1, la[cur ^ 1]); fillB(g + 1, lb[cur ^ 1]); }
        __syncthreads();
    }

    const int rbase = rowBase + w * 16 + ((lane & 16) ? 8 : 0);
    const int nc = lane & 15;
#pragma unroll
    for (int nt = 0; nt < NTB; ++nt) {
        int n = nt * 16 + nc;
        float bias = b1[n];
#pragma unroll
        for (int r = 0; r < 8; ++r) {
            float v = acc[nt][r] + bias;
            hid[(size_t)(rbase + r) * Pp + n] = (v > 0.0f ? v : 0.0f);
        }
    }
}

// ---------------------------------------------------------------------------
// Kernel 10: out = hidden @ out2_w^T + out2_b  (B,)
// ---------------------------------------------------------------------------
__global__ void k_out2(const float* __restrict__ hid, const float* __restrict__ w2,
                       const float* __restrict__ b2, float* __restrict__ out)
{
    int b = blockIdx.x * blockDim.x + threadIdx.x;
    if (b >= Bn) return;
    float a = b2[0];
    const float* hp = hid + (size_t)b * Pp;
    for (int p = 0; p < Pp; ++p) a += hp[p] * w2[p];
    out[b] = a;
}

// ---------------------------------------------------------------------------
// Host launcher
// ---------------------------------------------------------------------------
extern "C" void kernel_launch(void* const* d_in, const int* in_sizes, int n_in,
                              void* d_out, int out_size, void* d_ws, size_t ws_size,
                              hipStream_t stream)
{
    (void)in_sizes; (void)n_in; (void)out_size; (void)ws_size;
    const float* enc   = (const float*)d_in[0];
    const float* cw    = (const float*)d_in[1];
    const float* cb    = (const float*)d_in[2];
    const float* Wd    = (const float*)d_in[3];
    const float* Ud    = (const float*)d_in[4];
    const float* vd    = (const float*)d_in[5];
    const float* lwih  = (const float*)d_in[6];
    const float* lwhh  = (const float*)d_in[7];
    const float* lbih  = (const float*)d_in[8];
    const float* lbhh  = (const float*)d_in[9];
    const float* gwih  = (const float*)d_in[10];
    const float* gwhh  = (const float*)d_in[11];
    const float* gbih  = (const float*)d_in[12];
    const float* gbhh  = (const float*)d_in[13];
    const float* w1    = (const float*)d_in[14];
    const float* b1    = (const float*)d_in[15];
    const float* w2    = (const float*)d_in[16];
    const float* b2    = (const float*)d_in[17];
    float* out = (float*)d_out;

    // Workspace carve (256B-aligned)
    char* p = (char*)d_ws;
    auto carve = [&p](size_t bytes) -> void* {
        void* r = (void*)p;
        p += (bytes + 255) & ~(size_t)255;
        return r;
    };
    __bf16* Hb   = (__bf16*)carve((size_t)RND * Mm * 2);          // 77 MB (L2-resident)
    __bf16* HUb  = (__bf16*)carve((size_t)RND * Mm * 2);          // 77 MB (L2-resident)
    float*  dv   = (float*)carve((size_t)Bn * Pp * 4);
    float*  sv   = (float*)carve((size_t)Bn * Pp * 4);
    float*  qwd  = (float*)carve((size_t)Bn * Mm * 4);
    float*  cv   = (float*)carve((size_t)Bn * Mm * 4);
    float*  gts  = (float*)carve((size_t)Bn * 4 * Pp * 4);
    float*  hg   = (float*)carve((size_t)NRG * SHc * 4);
    float*  gg   = (float*)carve((size_t)NRG * 192 * 4);
    float*  hid  = (float*)carve((size_t)Bn * Pp * 4);

    // Zero recurrent state (harness does not re-zero between replays)
    hipMemsetAsync(dv, 0, (size_t)Bn * Pp * 4, stream);
    hipMemsetAsync(sv, 0, (size_t)Bn * Pp * 4, stream);
    hipMemsetAsync(hg, 0, (size_t)NRG * SHc * 4, stream);

    // Encoder: conv -> H (bf16), then HU (bf16); both stay hot in 192MB L2
    k_conv<<<dim3(RND / 128), dim3(256), 0, stream>>>(enc, cw, cb, Hb);
    k_hu  <<<dim3(RND / 128), dim3(256), 0, stream>>>(Hb, Ud, HUb);

    // Sequential attention+LSTM decoder (94 steps)
    for (int step = 0; step < TMc; ++step) {
        k_qwd  <<<dim3(Bn / 128), dim3(256), 0, stream>>>(dv, sv, Wd, qwd);
        k_attn <<<dim3(Bn), dim3(128), 0, stream>>>(qwd, HUb, Hb, vd, cv);
        k_gates<<<dim3(Bn / 128, 4), dim3(256), 0, stream>>>(cv, dv, lwih, lwhh, lbih, lbhh, gts);
        k_lstm <<<dim3((Bn * Pp + 255) / 256), dim3(256), 0, stream>>>(gts, dv, sv);
    }

    // Skip-GRU (9 steps over 40960 rows)
    for (int pt = 0; pt < 9; ++pt) {
        k_gru_gates<<<dim3(NRG / 128, 2), dim3(256), 0, stream>>>(Hb, hg, gwih, gwhh, gbih, gbhh, gg, pt);
        k_gru_upd  <<<dim3((NRG * SHc + 255) / 256), dim3(256), 0, stream>>>(gg, hg);
    }

    // Output MLP
    k_out1<<<dim3(Bn / 128), dim3(256), 0, stream>>>(dv, cv, hg, w1, b1, hid);
    k_out2<<<dim3((Bn + 255) / 256), dim3(256), 0, stream>>>(hid, w2, b2, out);
}
